// DeBERTaMoEClassifier_17248588661240
// MI455X (gfx1250) — matrix operand order repaired and checked by reading, not compiled, not run on hardware
//
#include <hip/hip_runtime.h>
#include <hip/hip_bf16.h>
#include <math.h>

// Problem constants (match the reference).
#define Bdim   1024
#define Sdim   128
#define Hdim   1024
#define Edim   16
#define HIDdim 1024
#define Cdim   3
#define EPSV   1e-5f

typedef unsigned short bfs; // raw bf16 storage
typedef __attribute__((ext_vector_type(16))) __bf16 v16bf;
typedef __attribute__((ext_vector_type(8)))  __bf16 v8bf;
typedef __attribute__((ext_vector_type(8)))  float  v8f;

__device__ __forceinline__ bfs f2bf(float f) {
  unsigned int u = __float_as_uint(f);
  u += 0x7FFFu + ((u >> 16) & 1u);      // round-to-nearest-even
  return (bfs)(u >> 16);
}

// CDNA5 async copy: 16B global -> LDS, tracked by ASYNCcnt (ISA 15.18.3 op 98).
// vdst = per-lane LDS byte address, vaddr = 64-bit global address, GV mode.
__device__ __forceinline__ void async_ld16(unsigned lds_addr, const bfs* g) {
  asm volatile("global_load_async_to_lds_b128 %0, %1, off"
               :: "v"(lds_addr), "v"((unsigned long long)g)
               : "memory");
}

// ---------------------------------------------------------------------------
// Kernel 1: stream hidden_states once -> cls (f32+bf16), m = mean_S (bf16)
// ---------------------------------------------------------------------------
__global__ __launch_bounds__(256) void reduce_cls_mean_k(
    const float* __restrict__ hs, float* __restrict__ clsf,
    bfs* __restrict__ clsb, bfs* __restrict__ mb)
{
  const int b = blockIdx.x, t = threadIdx.x;
  const float* base = hs + (long long)b * Sdim * Hdim;
  float acc[4] = {0.f, 0.f, 0.f, 0.f};
  for (int ss = 0; ss < Sdim; ++ss) {
    const float* row = base + (long long)ss * Hdim;
#pragma unroll
    for (int i = 0; i < 4; ++i) acc[i] += row[t + i * 256];
  }
#pragma unroll
  for (int i = 0; i < 4; ++i) {
    const int hh = t + i * 256;
    const float cv = base[hh];                       // s == 0 row
    clsf[(long long)b * Hdim + hh] = cv;
    clsb[(long long)b * Hdim + hh] = f2bf(cv);
    mb[(long long)b * Hdim + hh]   = f2bf(acc[i] * (1.f / Sdim));
  }
}

// ---------------------------------------------------------------------------
// Kernel 2: W (K x N) f32 row-major -> W^T (N x K) bf16 row-major, batched z.
// ---------------------------------------------------------------------------
__global__ __launch_bounds__(256) void convert_transpose_bf16(
    const float* __restrict__ in, bfs* __restrict__ out, int K, int N)
{
  __shared__ bfs tile[32][33];
  const long long zoff = (long long)blockIdx.z * K * N;
  in += zoff; out += zoff;
  const int n0 = blockIdx.x * 32, k0 = blockIdx.y * 32;
  const int tx = threadIdx.x, ty = threadIdx.y;
#pragma unroll
  for (int i = 0; i < 4; ++i)
    tile[ty + i * 8][tx] = f2bf(in[(long long)(k0 + ty + i * 8) * N + n0 + tx]);
  __syncthreads();
#pragma unroll
  for (int i = 0; i < 4; ++i)
    out[(long long)(n0 + ty + i * 8) * K + k0 + tx] = tile[tx][ty + i * 8];
}

// ---------------------------------------------------------------------------
// Batched WMMA GEMM:  C[z] = A[z] (MxK bf16 rowmajor) @ B[z]^T (NxK bf16
// rowmajor) + bias[z][col].
//   Block tile 64x128, 128 threads (4 waves). Each wave owns a 32x64 strip =
//   2x4 v_wmma_f32_16x16x32_bf16 tiles. Double-buffered LDS stages filled by
//   async global->LDS (ASYNCcnt), k-step = 32.
// ---------------------------------------------------------------------------
__device__ __forceinline__ v16bf pack_frag(const bfs* p, int hiOff) {
  const v8bf lo = *(const v8bf*)p;
  const v8bf hi = *(const v8bf*)(p + hiOff);
  v16bf o;
#pragma unroll
  for (int i = 0; i < 8; ++i) { o[i] = lo[i]; o[i + 8] = hi[i]; }
  return o;
}

__device__ __forceinline__ void store_tile16(const v8f& acc, float* C, int ldc,
                                             int row0, int col0,
                                             const float* bias)
{
  const int lane = threadIdx.x & 31;
  const int col = col0 + (lane & 15);
  const int row = row0 + (lane >> 4) * 8;   // C/D: VGPR i -> rows i and 8+i
  const float bv = bias[col];
#pragma unroll
  for (int i = 0; i < 8; ++i)
    C[(long long)(row + i) * ldc + col] = acc[i] + bv;
}

__global__ __launch_bounds__(128) void gemm_bf16_tn(
    const bfs* __restrict__ A, const bfs* __restrict__ Bt,
    const float* __restrict__ bias, float* __restrict__ C,
    int M, int N, int Kd, int lda, int ldb, int ldc,
    long long sA, long long sB, long long sC, long long sBias)
{
  (void)M; (void)N;
  // pitch 40 halves (80B, 16B-aligned rows, spreads banks)
  __shared__ bfs As[2][64 * 40];
  __shared__ bfs Bs[2][128 * 40];

  const int tid  = threadIdx.x;
  const int lane = tid & 31;
  const int wave = tid >> 5;
  const int wr = (wave & 1) * 32;    // wave row strip within 64
  const int wc = (wave >> 1) * 64;   // wave col strip within 128

  const long long z = blockIdx.z;
  A += z * sA; Bt += z * sB; C += z * sC;
  const float* bp = bias + z * sBias;

  const int tM = blockIdx.y * 64;
  const int tN = blockIdx.x * 128;

  const int lr = tid >> 2;          // 0..31 : row within 32-row group
  const int lc = (tid & 3) * 8;     // 0,8,16,24 halves : 16B k-chunk

  const unsigned asA[2] = { (unsigned)(size_t)(&As[0][0]),
                            (unsigned)(size_t)(&As[1][0]) };
  const unsigned asB[2] = { (unsigned)(size_t)(&Bs[0][0]),
                            (unsigned)(size_t)(&Bs[1][0]) };

  // Issue the 6 async 16B transfers of one K-stage (A: 2 rows, B: 4 rows).
  auto issue_stage = [&](int k0, int buf) {
    const unsigned aBase = asA[buf] + (unsigned)((lr * 40 + lc) * 2);
    const unsigned bBase = asB[buf] + (unsigned)((lr * 40 + lc) * 2);
    const bfs* ag = A  + (long long)(tM + lr) * lda + k0 + lc;
    const bfs* bg = Bt + (long long)(tN + lr) * ldb + k0 + lc;
    async_ld16(aBase,               ag);
    async_ld16(aBase + 32 * 40 * 2, ag + (long long)32 * lda);
    async_ld16(bBase,               bg);
    async_ld16(bBase + 32 * 40 * 2, bg + (long long)32 * ldb);
    async_ld16(bBase + 64 * 40 * 2, bg + (long long)64 * ldb);
    async_ld16(bBase + 96 * 40 * 2, bg + (long long)96 * ldb);
  };

  v8f acc[2][4];
#pragma unroll
  for (int i = 0; i < 2; ++i)
#pragma unroll
    for (int j = 0; j < 4; ++j) {
      v8f zv = {0,0,0,0,0,0,0,0};
      acc[i][j] = zv;
    }

  const int nK = Kd / 32;
  issue_stage(0, 0);

  for (int ks = 0; ks < nK; ++ks) {
    const int buf = ks & 1;
    if (ks + 1 < nK) {
      issue_stage((ks + 1) * 32, buf ^ 1);
      // 6 newer transfers may stay in flight; drain the 6 of stage `ks`.
      asm volatile("s_wait_asynccnt 0x6" ::: "memory");
    } else {
      asm volatile("s_wait_asynccnt 0x0" ::: "memory");
    }
    __syncthreads();   // stage `ks` visible to all waves; prev compute done

    const bfs* as = &As[buf][0];
    const bfs* bs = &Bs[buf][0];
    const int fr = lane & 15;

    // A frag (16x32): lane m<16 holds K{0..7,16..23}; lane m+16 K{8..15,24..31}
    const int ak = (lane >> 4) * 8;
    const bfs* pa = as + (wr + fr) * 40 + ak;
    v16bf fa[2];
    fa[0] = pack_frag(pa, 16);
    fa[1] = pack_frag(pa + 16 * 40, 16);

    // B frag (32x16): lane n<16 holds col n K{0..7,16..23}; lane n+16 K{8..15,24..31}
    const int bk = (lane >> 4) * 8;
    v16bf fb[4];
#pragma unroll
    for (int j = 0; j < 4; ++j)
      fb[j] = pack_frag(bs + (wc + j * 16 + fr) * 40 + bk, 16);

#pragma unroll
    for (int i = 0; i < 2; ++i)
#pragma unroll
      for (int j = 0; j < 4; ++j)
        acc[i][j] = __builtin_amdgcn_wmma_f32_16x16x32_bf16(
            false, fa[i], false, fb[j], (short)0, acc[i][j], false, false);

    __syncthreads();   // protect buf before it is refilled next iteration
  }

#pragma unroll
  for (int i = 0; i < 2; ++i)
#pragma unroll
    for (int j = 0; j < 4; ++j)
      store_tile16(acc[i][j], C, ldc, tM + wr + 16 * i, tN + wc + 16 * j, bp);
}

// ---------------------------------------------------------------------------
// Router: logits = cls@Wr + br, top-4 softmax scattered into rw (B,16).
// ---------------------------------------------------------------------------
__global__ __launch_bounds__(256) void router_topk_k(
    const float* __restrict__ clsf, const float* __restrict__ Wr,
    const float* __restrict__ br, float* __restrict__ rw)
{
  __shared__ float part[256];
  const int b = blockIdx.x, t = threadIdx.x;
  const int e = t & 15, kg = t >> 4;
  float acc = 0.f;
  for (int k = kg; k < Hdim; k += 16)
    acc += clsf[(long long)b * Hdim + k] * Wr[(long long)k * Edim + e];
  part[t] = acc; __syncthreads();
  for (int sft = 128; sft >= 16; sft >>= 1) {
    if (t < sft) part[t] += part[t + sft];
    __syncthreads();
  }
  if (t == 0) {
    float v[16], out[16];
    for (int i = 0; i < 16; ++i) { v[i] = part[i] + br[i]; out[i] = 0.f; }
    float tv[4]; int ti[4];
    for (int j = 0; j < 4; ++j) {
      int bi = 0; float bv = v[0];
      for (int i = 1; i < 16; ++i) if (v[i] > bv) { bv = v[i]; bi = i; }
      tv[j] = bv; ti[j] = bi; v[bi] = -3.4e38f;
    }
    const float mx = tv[0];
    float se = 0.f, p[4];
    for (int j = 0; j < 4; ++j) { p[j] = expf(tv[j] - mx); se += p[j]; }
    for (int j = 0; j < 4; ++j) out[ti[j]] = p[j] / se;
    for (int i = 0; i < 16; ++i) rw[(long long)b * Edim + i] = out[i];
  }
}

// ---------------------------------------------------------------------------
// original_logits = tanh(t1) @ Wo + bo
// ---------------------------------------------------------------------------
__global__ __launch_bounds__(256) void orig_logits_k(
    const float* __restrict__ t1, const float* __restrict__ Wo,
    const float* __restrict__ bo, float* __restrict__ orig)
{
  __shared__ float r0[256], r1[256], r2[256];
  const int b = blockIdx.x, t = threadIdx.x;
  float a0 = 0.f, a1 = 0.f, a2 = 0.f;
  for (int hh = t; hh < Hdim; hh += 256) {
    const float x = tanhf(t1[(long long)b * Hdim + hh]);
    a0 += x * Wo[hh * 3 + 0]; a1 += x * Wo[hh * 3 + 1]; a2 += x * Wo[hh * 3 + 2];
  }
  r0[t] = a0; r1[t] = a1; r2[t] = a2; __syncthreads();
  for (int sft = 128; sft > 0; sft >>= 1) {
    if (t < sft) { r0[t] += r0[t + sft]; r1[t] += r1[t + sft]; r2[t] += r2[t + sft]; }
    __syncthreads();
  }
  if (t == 0) {
    orig[(long long)b * 3 + 0] = r0[0] + bo[0];
    orig[(long long)b * 3 + 1] = r1[0] + bo[1];
    orig[(long long)b * 3 + 2] = r2[0] + bo[2];
  }
}

// ---------------------------------------------------------------------------
// LayerNorm (per (b,e) over HID) + exact GELU -> bf16
// ---------------------------------------------------------------------------
__global__ __launch_bounds__(256) void ln_gelu_k(
    const float* __restrict__ h, const float* __restrict__ g1,
    const float* __restrict__ beta1, bfs* __restrict__ hg)
{
  __shared__ float r1[256], r2[256];
  __shared__ float smu, srs;
  const long long be = blockIdx.x;
  const int e = (int)(be % Edim);
  const int t = threadIdx.x;
  const float* x = h + be * HIDdim;
  float v[4]; float s = 0.f, s2 = 0.f;
#pragma unroll
  for (int i = 0; i < 4; ++i) { v[i] = x[t + i * 256]; s += v[i]; s2 += v[i] * v[i]; }
  r1[t] = s; r2[t] = s2; __syncthreads();
  for (int sft = 128; sft > 0; sft >>= 1) {
    if (t < sft) { r1[t] += r1[t + sft]; r2[t] += r2[t + sft]; }
    __syncthreads();
  }
  if (t == 0) {
    const float mu = r1[0] * (1.f / HIDdim);
    const float var = r2[0] * (1.f / HIDdim) - mu * mu;
    smu = mu; srs = rsqrtf(var + EPSV);
  }
  __syncthreads();
#pragma unroll
  for (int i = 0; i < 4; ++i) {
    const int d = t + i * 256;
    const float xn = (v[i] - smu) * srs;
    const float y = xn * g1[(long long)e * HIDdim + d] + beta1[(long long)e * HIDdim + d];
    const float ge = 0.5f * y * (1.f + erff(y * 0.70710678118654752f));
    hg[be * HIDdim + d] = f2bf(ge);
  }
}

// ---------------------------------------------------------------------------
// expert_logits[b,e,:] = h2[b,e,:] @ Wp + bp
// ---------------------------------------------------------------------------
__global__ __launch_bounds__(256) void expert_proj_k(
    const float* __restrict__ h2, const float* __restrict__ Wp,
    const float* __restrict__ bpv, float* __restrict__ el)
{
  __shared__ float r0[256], r1[256], r2[256];
  const long long be = blockIdx.x;
  const int t = threadIdx.x;
  const float* x = h2 + be * HIDdim;
  float a0 = 0.f, a1 = 0.f, a2 = 0.f;
  for (int d = t; d < HIDdim; d += 256) {
    const float xv = x[d];
    a0 += xv * Wp[d * 3 + 0]; a1 += xv * Wp[d * 3 + 1]; a2 += xv * Wp[d * 3 + 2];
  }
  r0[t] = a0; r1[t] = a1; r2[t] = a2; __syncthreads();
  for (int sft = 128; sft > 0; sft >>= 1) {
    if (t < sft) { r0[t] += r0[t + sft]; r1[t] += r1[t + sft]; r2[t] += r2[t + sft]; }
    __syncthreads();
  }
  if (t == 0) {
    el[be * 3 + 0] = r0[0] + bpv[0];
    el[be * 3 + 1] = r1[0] + bpv[1];
    el[be * 3 + 2] = r2[0] + bpv[2];
  }
}

// ---------------------------------------------------------------------------
// Final head: moe combine + concat + Wf1 + LN(3) + relu + Wf2. 1 thread / b.
// ---------------------------------------------------------------------------
__global__ __launch_bounds__(256) void final_head_k(
    const float* __restrict__ orig, const float* __restrict__ rw,
    const float* __restrict__ el,
    const float* __restrict__ Wf1, const float* __restrict__ bf1,
    const float* __restrict__ gf, const float* __restrict__ betaf,
    const float* __restrict__ Wf2, const float* __restrict__ bf2,
    float* __restrict__ outp)
{
  const int b = blockIdx.x * 256 + threadIdx.x;
  if (b >= Bdim) return;
  float comb[6];
  for (int c = 0; c < 3; ++c) comb[c] = orig[(long long)b * 3 + c];
  float moe[3] = {0.f, 0.f, 0.f};
  for (int e = 0; e < Edim; ++e) {
    const float w = rw[(long long)b * Edim + e];
    const float* ep = el + ((long long)b * Edim + e) * 3;
    moe[0] += w * ep[0]; moe[1] += w * ep[1]; moe[2] += w * ep[2];
  }
  comb[3] = moe[0]; comb[4] = moe[1]; comb[5] = moe[2];
  float y[3];
  for (int c = 0; c < 3; ++c) {
    float a = bf1[c];
    for (int j = 0; j < 6; ++j) a += comb[j] * Wf1[j * 3 + c];
    y[c] = a;
  }
  const float mu = (y[0] + y[1] + y[2]) * (1.f / 3.f);
  const float var = ((y[0]-mu)*(y[0]-mu) + (y[1]-mu)*(y[1]-mu) + (y[2]-mu)*(y[2]-mu)) * (1.f / 3.f);
  const float rs = rsqrtf(var + EPSV);
  float r[3];
  for (int c = 0; c < 3; ++c) {
    const float z = (y[c] - mu) * rs * gf[c] + betaf[c];
    r[c] = z > 0.f ? z : 0.f;
  }
  for (int c = 0; c < 3; ++c) {
    float a = bf2[c];
    for (int j = 0; j < 3; ++j) a += r[j] * Wf2[j * 3 + c];
    outp[(long long)b * 3 + c] = a;
  }
}

// ---------------------------------------------------------------------------
extern "C" void kernel_launch(void* const* d_in, const int* in_sizes, int n_in,
                              void* d_out, int out_size, void* d_ws, size_t ws_size,
                              hipStream_t stream)
{
  (void)in_sizes; (void)n_in; (void)out_size; (void)ws_size;
  const float* hs    = (const float*)d_in[0];
  const float* Wd    = (const float*)d_in[1];
  const float* bd    = (const float*)d_in[2];
  const float* Wo    = (const float*)d_in[3];
  const float* bo    = (const float*)d_in[4];
  const float* Wr    = (const float*)d_in[5];
  const float* br    = (const float*)d_in[6];
  const float* We1   = (const float*)d_in[7];
  const float* be1   = (const float*)d_in[8];
  const float* g1    = (const float*)d_in[9];
  const float* beta1 = (const float*)d_in[10];
  const float* We2   = (const float*)d_in[11];
  const float* be2   = (const float*)d_in[12];
  const float* Wp    = (const float*)d_in[13];
  const float* bp    = (const float*)d_in[14];
  const float* Wf1   = (const float*)d_in[15];
  const float* bf1   = (const float*)d_in[16];
  const float* gf    = (const float*)d_in[17];
  const float* betaf = (const float*)d_in[18];
  const float* Wf2   = (const float*)d_in[19];
  const float* bf2   = (const float*)d_in[20];
  float* outp = (float*)d_out;

  // Workspace arena (256B-aligned carve-outs).
  char* wsb = (char*)d_ws;
  size_t off = 0;
  auto carve = [&](size_t bytes) -> void* {
    off = (off + 255) & ~(size_t)255;
    void* p = wsb + off;
    off += bytes;
    return p;
  };
  float* clsf = (float*)carve((size_t)Bdim * Hdim * 4);
  bfs*   clsb = (bfs*)  carve((size_t)Bdim * Hdim * 2);
  bfs*   mb   = (bfs*)  carve((size_t)Bdim * Hdim * 2);
  bfs*   WdT  = (bfs*)  carve((size_t)Hdim * Hdim * 2);
  bfs*   We1T = (bfs*)  carve((size_t)Edim * Hdim * HIDdim * 2);
  bfs*   We2T = (bfs*)  carve((size_t)Edim * HIDdim * HIDdim * 2);
  float* t1   = (float*)carve((size_t)Bdim * Hdim * 4);
  float* h    = (float*)carve((size_t)Bdim * Edim * HIDdim * 4); // reused as h2
  bfs*   hg   = (bfs*)  carve((size_t)Bdim * Edim * HIDdim * 2);
  float* rw   = (float*)carve((size_t)Bdim * Edim * 4);
  float* orig = (float*)carve((size_t)Bdim * 3 * 4);
  float* el   = (float*)carve((size_t)Bdim * Edim * 3 * 4);
  float* h2   = h;   // h (f32) dead after ln_gelu produced hg

  // 1) Stream hidden_states once.
  reduce_cls_mean_k<<<Bdim, 256, 0, stream>>>(hs, clsf, clsb, mb);

  // 2) Transpose-convert weights to bf16 (NxK) for the WMMA B operand.
  convert_transpose_bf16<<<dim3(Hdim/32, Hdim/32, 1),        dim3(32, 8), 0, stream>>>(Wd,  WdT,  Hdim,   Hdim);
  convert_transpose_bf16<<<dim3(HIDdim/32, Hdim/32, Edim),   dim3(32, 8), 0, stream>>>(We1, We1T, Hdim,   HIDdim);
  convert_transpose_bf16<<<dim3(HIDdim/32, HIDdim/32, Edim), dim3(32, 8), 0, stream>>>(We2, We2T, HIDdim, HIDdim);

  // 3) t1 = cls @ Wd + bd   (tanh fused into orig_logits_k)
  gemm_bf16_tn<<<dim3(Hdim/128, Bdim/64, 1), 128, 0, stream>>>(
      clsb, WdT, bd, t1, Bdim, Hdim, Hdim,
      Hdim, Hdim, Hdim, 0LL, 0LL, 0LL, 0LL);

  // 4) Router (reads cls f32) and original logits (reads t1).
  router_topk_k<<<Bdim, 256, 0, stream>>>(clsf, Wr, br, rw);
  orig_logits_k<<<Bdim, 256, 0, stream>>>(t1, Wo, bo, orig);

  // 5) h[b,e,:] = m @ We1[e] + be1[e]   (batched over experts)
  gemm_bf16_tn<<<dim3(HIDdim/128, Bdim/64, Edim), 128, 0, stream>>>(
      mb, We1T, be1, h, Bdim, HIDdim, Hdim,
      Hdim, Hdim, Edim * HIDdim,
      0LL, (long long)Hdim * HIDdim, (long long)HIDdim, (long long)HIDdim);

  // 6) LN + GELU -> bf16
  ln_gelu_k<<<Bdim * Edim, 256, 0, stream>>>(h, g1, beta1, hg);

  // 7) h2[b,e,:] = hg @ We2[e] + be2[e]
  gemm_bf16_tn<<<dim3(HIDdim/128, Bdim/64, Edim), 128, 0, stream>>>(
      hg, We2T, be2, h2, Bdim, HIDdim, HIDdim,
      Edim * HIDdim, HIDdim, Edim * HIDdim,
      (long long)HIDdim, (long long)HIDdim * HIDdim, (long long)HIDdim, (long long)HIDdim);

  // 8) expert logits, then final head.
  expert_proj_k<<<Bdim * Edim, 256, 0, stream>>>(h2, Wp, bp, el);
  final_head_k<<<(Bdim + 255) / 256, 256, 0, stream>>>(
      orig, rw, el, Wf1, bf1, gf, betaf, Wf2, bf2, outp);
}
